// ServicePredictorNNSetReprHydra_80839874445315
// MI455X (gfx1250) — matrix operand order repaired and checked by reading, not compiled, not run on hardware
//
#include <hip/hip_runtime.h>
#include <hip/hip_bf16.h>
#include <math.h>

typedef __attribute__((ext_vector_type(16))) _Float16 v16h;
typedef __attribute__((ext_vector_type(8)))  _Float16 v8h;
typedef __attribute__((ext_vector_type(8)))  float    v8f;

#define B_SZ 8192
#define L_SZ 512
#define NE   16
#define ED   5
#define F1   7     // ED + 2 loads
#define H1   50
#define H2   25
#define FXO  5
#define NTGT 3
#define NLM  4

// padded (WMMA) dims
#define F1P  32
#define H1P  64
#define H2P  32
#define FXOP 16

#define TILE_STRIDE 72   // halves per staging row (16-byte aligned, bank-skewed)

__device__ __forceinline__ float gelu_exact(float x) {
    return 0.5f * x * (1.0f + erff(x * 0.70710678118654752f));
}

union V16 { v16h v; v8h h[2]; _Float16 e[16]; };

__device__ __forceinline__ v8f wmma_f16(v16h a, v16h b, v8f c) {
    return __builtin_amdgcn_wmma_f32_16x16x32_f16(
        /*neg_a=*/false, a, /*neg_b=*/false, b,
        /*c_mod=*/(short)0, c, /*reuse_a=*/false, /*reuse_b=*/false);
}

// ---------------------------------------------------------------------------
// Kernel 1: per-(b,l) fx MLP on WMMA + masked length-sum -> h1 (B,5)
// grid = B blocks, 256 threads (8 waves); wave handles 4 tiles of 16 positions
// ---------------------------------------------------------------------------
__global__ __launch_bounds__(256) void fx_kernel(
    const float* __restrict__ features,   // (B,3,L)
    const int*   __restrict__ lengths,    // (B,)
    const float* __restrict__ emb_table,  // (16,5)
    const float* __restrict__ w1, const float* __restrict__ b1,   // (7,50),(50)
    const float* __restrict__ w2, const float* __restrict__ b2,   // (50,25),(25)
    const float* __restrict__ w3, const float* __restrict__ b3,   // (25,5),(5)
    float* __restrict__ h1_out)           // (B,5)
{
    __shared__ _Float16 sW1T[H1P * F1P];    // [n][k], k stride 32
    __shared__ _Float16 sW2T[H2P * H1P];    // [n][k], k stride 64
    __shared__ _Float16 sW3T[FXOP * F1P];   // [n][k], k stride 32
    __shared__ float sB1[H1P], sB2[H2P], sB3[FXOP];
    __shared__ float sEmb[NE * ED];
    __shared__ _Float16 sTile[8][16 * TILE_STRIDE];  // per-wave activation staging
    __shared__ float sH1[FXO];

    const int tid = threadIdx.x;
    const int b   = blockIdx.x;

    // ---- stage transposed, zero-padded weights into LDS ----
    for (int i = tid; i < H1P * F1P; i += 256) {
        int n = i >> 5, k = i & 31;
        sW1T[i] = (_Float16)((k < F1 && n < H1) ? w1[k * H1 + n] : 0.f);
    }
    for (int i = tid; i < H2P * H1P; i += 256) {
        int n = i >> 6, k = i & 63;
        sW2T[i] = (_Float16)((k < H1 && n < H2) ? w2[k * H2 + n] : 0.f);
    }
    for (int i = tid; i < FXOP * F1P; i += 256) {
        int n = i >> 5, k = i & 31;
        sW3T[i] = (_Float16)((k < H2 && n < FXO) ? w3[k * FXO + n] : 0.f);
    }
    if (tid < H1P)     sB1[tid] = (tid < H1)  ? b1[tid] : 0.f;
    if (tid < H2P)     sB2[tid] = (tid < H2)  ? b2[tid] : 0.f;
    if (tid < FXOP)    sB3[tid] = (tid < FXO) ? b3[tid] : 0.f;
    if (tid < NE * ED) sEmb[tid] = emb_table[tid];
    if (tid < FXO)     sH1[tid]  = 0.f;
    __syncthreads();

    const int  wave  = tid >> 5;
    const int  lane  = tid & 31;
    const int  nlo   = lane & 15;
    const bool hilane = lane >= 16;
    const int  kbase = hilane ? 16 : 0;   // B-frag K base for this lane half
    const int  mofs  = hilane ? 8 : 0;    // C-frag row base for this lane half
    _Float16* tile = sTile[wave];
    const int len = lengths[b];

    // ---- hoist B fragments (constant across all tiles) ----
    v16h B1f[4], B2f[2][2], B3f;
#pragma unroll
    for (int nt = 0; nt < 4; ++nt)
        B1f[nt] = *(const v16h*)&sW1T[(nt * 16 + nlo) * F1P + kbase];
#pragma unroll
    for (int nt = 0; nt < 2; ++nt)
#pragma unroll
        for (int kc = 0; kc < 2; ++kc)
            B2f[nt][kc] = *(const v16h*)&sW2T[(nt * 16 + nlo) * H1P + kc * 32 + kbase];
    B3f = *(const v16h*)&sW3T[nlo * F1P + kbase];

    const float* idrow = features + (size_t)b * 3 * L_SZ;
    const float* ld0   = idrow + L_SZ;
    const float* ld1   = idrow + 2 * L_SZ;

    float acc = 0.f;   // partial h1 for output column nlo (valid when nlo < FXO)

    for (int t = 0; t < 4; ++t) {
        const int l0 = (wave * 4 + t) * 16;

        // ---- build layer-1 A fragment (K=7 real of 32) ----
        V16 A1;
#pragma unroll
        for (int i = 0; i < 16; ++i) A1.e[i] = (_Float16)0.f;
        if (!hilane) {
            const int l  = l0 + nlo;
            const int id = ((int)idrow[l]) & (NE - 1);
#pragma unroll
            for (int e = 0; e < ED; ++e) A1.e[e] = (_Float16)sEmb[id * ED + e];
            A1.e[5] = (_Float16)ld0[l];
            A1.e[6] = (_Float16)ld1[l];
        }

        // ---- layer 1: gelu(A1 @ W1 + b1) -> LDS tile (cols 0..63) ----
#pragma unroll
        for (int nt = 0; nt < 4; ++nt) {
            const float bv = sB1[nt * 16 + nlo];
            v8f c = {bv, bv, bv, bv, bv, bv, bv, bv};
            c = wmma_f16(A1.v, B1f[nt], c);
#pragma unroll
            for (int r = 0; r < 8; ++r)
                tile[(r + mofs) * TILE_STRIDE + nt * 16 + nlo] =
                    (_Float16)gelu_exact(c[r]);
        }

        // ---- layer 2: load A frags (both K-chunks) before overwriting ----
        V16 A2[2];
#pragma unroll
        for (int kc = 0; kc < 2; ++kc) {
            const int c0 = kc * 32 + (hilane ? 16 : 0);
            A2[kc].h[0] = *(const v8h*)&tile[nlo * TILE_STRIDE + c0];
            A2[kc].h[1] = *(const v8h*)&tile[nlo * TILE_STRIDE + c0 + 8];
        }
#pragma unroll
        for (int nt = 0; nt < 2; ++nt) {
            const float bv = sB2[nt * 16 + nlo];
            v8f c = {bv, bv, bv, bv, bv, bv, bv, bv};
            c = wmma_f16(A2[0].v, B2f[nt][0], c);
            c = wmma_f16(A2[1].v, B2f[nt][1], c);
#pragma unroll
            for (int r = 0; r < 8; ++r)
                tile[(r + mofs) * TILE_STRIDE + nt * 16 + nlo] =
                    (_Float16)gelu_exact(c[r]);
        }

        // ---- layer 3 + masked reduction over positions ----
        {
            V16 A3;
            const int c0 = hilane ? 16 : 0;
            A3.h[0] = *(const v8h*)&tile[nlo * TILE_STRIDE + c0];
            A3.h[1] = *(const v8h*)&tile[nlo * TILE_STRIDE + c0 + 8];
            const float bv = sB3[nlo];
            v8f c = {bv, bv, bv, bv, bv, bv, bv, bv};
            c = wmma_f16(A3.v, B3f, c);
#pragma unroll
            for (int r = 0; r < 8; ++r) {
                const int l = l0 + r + mofs;
                acc += (l < len) ? c[r] : 0.f;
            }
        }
    }

    if (nlo < FXO) atomicAdd(&sH1[nlo], acc);   // both lane halves contribute
    __syncthreads();
    if (tid < FXO) h1_out[(size_t)b * FXO + tid] = sH1[tid];
}

// ---------------------------------------------------------------------------
// Kernel 2: tiny rg/lm heads, one thread per batch row (~46 MFLOP total)
// ---------------------------------------------------------------------------
__global__ __launch_bounds__(256) void heads_kernel(
    const float* __restrict__ x_features,  // (B,3)
    const float* __restrict__ emb_table,   // (16,5)
    const float* __restrict__ h1,          // (B,5)
    const float* __restrict__ rw1, const float* __restrict__ rb1,  // (12,50)
    const float* __restrict__ rw2, const float* __restrict__ rb2,  // (50,25)
    const float* __restrict__ rw3, const float* __restrict__ rb3,  // (25,3)
    const float* __restrict__ lw1, const float* __restrict__ lb1,  // (7,50)
    const float* __restrict__ lw2, const float* __restrict__ lb2,  // (50,10)
    const float* __restrict__ lw3, const float* __restrict__ lb3,  // (10,4)
    float* __restrict__ out_sts,   // (B,3)
    float* __restrict__ out_lms)   // (B,4)
{
    const int b = blockIdx.x * blockDim.x + threadIdx.x;
    if (b >= B_SZ) return;

    float xf[F1];
    {
        const int id = ((int)x_features[b * 3]) & (NE - 1);
#pragma unroll
        for (int e = 0; e < ED; ++e) xf[e] = emb_table[id * ED + e];
        xf[5] = x_features[b * 3 + 1];
        xf[6] = x_features[b * 3 + 2];
    }

    // ---- rg head: concat(h1, xf) 12 -> 50 -> 25 -> 3, 10*tanh ----
    float in12[FXO + F1];
#pragma unroll
    for (int i = 0; i < FXO; ++i) in12[i] = h1[(size_t)b * FXO + i];
#pragma unroll
    for (int i = 0; i < F1; ++i) in12[FXO + i] = xf[i];

    float t1[H1];
    for (int j = 0; j < H1; ++j) {
        float s = rb1[j];
        for (int i = 0; i < FXO + F1; ++i) s += in12[i] * rw1[i * H1 + j];
        t1[j] = gelu_exact(s);
    }
    float t2[H2];
    for (int j = 0; j < H2; ++j) {
        float s = rb2[j];
        for (int i = 0; i < H1; ++i) s += t1[i] * rw2[i * H2 + j];
        t2[j] = gelu_exact(s);
    }
    for (int j = 0; j < NTGT; ++j) {
        float s = rb3[j];
        for (int i = 0; i < H2; ++i) s += t2[i] * rw3[i * NTGT + j];
        out_sts[(size_t)b * NTGT + j] = 10.0f * tanhf(s);
    }

    // ---- lm head: xf 7 -> 50 -> 10 -> 4, gelu output ----
    float u1[H1];
    for (int j = 0; j < H1; ++j) {
        float s = lb1[j];
        for (int i = 0; i < F1; ++i) s += xf[i] * lw1[i * H1 + j];
        u1[j] = gelu_exact(s);
    }
    float u2[10];
    for (int j = 0; j < 10; ++j) {
        float s = lb2[j];
        for (int i = 0; i < H1; ++i) s += u1[i] * lw2[i * 10 + j];
        u2[j] = gelu_exact(s);
    }
    for (int j = 0; j < NLM; ++j) {
        float s = lb3[j];
        for (int i = 0; i < 10; ++i) s += u2[i] * lw3[i * NLM + j];
        out_lms[(size_t)b * NLM + j] = gelu_exact(s);
    }
}

extern "C" void kernel_launch(void* const* d_in, const int* in_sizes, int n_in,
                              void* d_out, int out_size, void* d_ws, size_t ws_size,
                              hipStream_t stream) {
    const float* features   = (const float*)d_in[0];
    const float* x_features = (const float*)d_in[1];
    const int*   lengths    = (const int*)d_in[2];
    const float* emb_table  = (const float*)d_in[3];
    const float* fx_w1 = (const float*)d_in[4];
    const float* fx_b1 = (const float*)d_in[5];
    const float* fx_w2 = (const float*)d_in[6];
    const float* fx_b2 = (const float*)d_in[7];
    const float* fx_w3 = (const float*)d_in[8];
    const float* fx_b3 = (const float*)d_in[9];
    const float* rg_w1 = (const float*)d_in[10];
    const float* rg_b1 = (const float*)d_in[11];
    const float* rg_w2 = (const float*)d_in[12];
    const float* rg_b2 = (const float*)d_in[13];
    const float* rg_w3 = (const float*)d_in[14];
    const float* rg_b3 = (const float*)d_in[15];
    const float* lm_w1 = (const float*)d_in[16];
    const float* lm_b1 = (const float*)d_in[17];
    const float* lm_w2 = (const float*)d_in[18];
    const float* lm_b2 = (const float*)d_in[19];
    const float* lm_w3 = (const float*)d_in[20];
    const float* lm_b3 = (const float*)d_in[21];

    float* h1  = (float*)d_ws;                     // (B,5) scratch
    float* out_sts = (float*)d_out;                // (B,3)
    float* out_lms = (float*)d_out + (size_t)B_SZ * NTGT;  // (B,4)

    fx_kernel<<<B_SZ, 256, 0, stream>>>(features, lengths, emb_table,
                                        fx_w1, fx_b1, fx_w2, fx_b2, fx_w3, fx_b3,
                                        h1);
    heads_kernel<<<(B_SZ + 255) / 256, 256, 0, stream>>>(
        x_features, emb_table, h1,
        rg_w1, rg_b1, rg_w2, rg_b2, rg_w3, rg_b3,
        lm_w1, lm_b1, lm_w2, lm_b2, lm_w3, lm_b3,
        out_sts, out_lms);
}
